// SE3Transformer_5428838662300
// MI455X (gfx1250) — compile-verified, matrix operand order checked by reading, not run on hardware
//
#include <hip/hip_runtime.h>
#include <hip/hip_bf16.h>
#include <math.h>

#define N_NODES 10000
#define N_EDGES 320000
#define HID 32

typedef _Float16 f16;
typedef f16   v16h __attribute__((ext_vector_type(16)));
typedef float v8f  __attribute__((ext_vector_type(8)));

__host__ __device__ constexpr int imin3(int a,int b){return a<b?a:b;}
__host__ __device__ constexpr int iabs3(int a){return a<0?-a:a;}
__host__ __device__ constexpr int MULTD(int d){ return d==0?4:(d==1?2:1); }

// flat offset of the J-tables for ordered pair (di,do) inside the Q table blob
__host__ __device__ constexpr int qpairOffset(int di,int doo){
  int off=0;
  for(int a=0;a<3;a++) for(int b=0;b<3;b++){
    if(a==di && b==doo) return off;
    int jmin=iabs3(a-b), nj=2*imin3(a,b)+1;
    for(int t=0;t<nj;t++){ int J=jmin+t; off += (2*J+1)*(2*b+1)*(2*a+1); }
  }
  return off;
}

// ------------------------- CG table generation -------------------------
struct cplx { double re, im; };
__device__ inline cplx cmul(cplx a, cplx b){ return { a.re*b.re - a.im*b.im, a.re*b.im + a.im*b.re }; }
__device__ inline cplx conjc(cplx a){ return { a.re, -a.im }; }
__device__ double dfact(int n){ double r=1.0; for(int i=2;i<=n;i++) r*=(double)i; return r; }

__device__ double su2cg(int j1,int m1,int j2,int m2,int j3,int m3){
  if(m1+m2!=m3) return 0.0;
  int vmin = 0; if(j2-j3-m1>vmin) vmin=j2-j3-m1; if(j1-j3+m2>vmin) vmin=j1-j3+m2;
  int vmax = j1+j2-j3; if(j1-m1<vmax) vmax=j1-m1; if(j2+m2<vmax) vmax=j2+m2;
  if(vmax<vmin) return 0.0;
  double pref = sqrt((2.0*j3+1.0)*dfact(j3+j1-j2)*dfact(j3-j1+j2)*dfact(j1+j2-j3)/dfact(j1+j2+j3+1));
  pref *= sqrt(dfact(j3+m3)*dfact(j3-m3)*dfact(j1-m1)*dfact(j1+m1)*dfact(j2-m2)*dfact(j2+m2));
  double s=0.0;
  for(int v=vmin;v<=vmax;v++){
    double d = dfact(v)*dfact(j1+j2-j3-v)*dfact(j1-m1-v)*dfact(j2+m2-v)*dfact(j3-j2+m1+v)*dfact(j3-j1-m2+v);
    s += ((v&1)? -1.0:1.0)/d;
  }
  return pref*s;
}

__device__ void r2c_fill(int l, cplx* q){     // q row-major (2l+1)x(2l+1)
  int n=2*l+1;
  for(int i=0;i<n*n;i++) q[i]={0.0,0.0};
  double inv = 1.0/sqrt(2.0);
  for(int m=-l;m<0;m++){
    q[(l+m)*n + (l-m)] = { inv, 0.0 };
    q[(l+m)*n + (l+m)] = { 0.0, -inv };
  }
  q[l*n+l] = {1.0,0.0};
  for(int m=1;m<=l;m++){
    double s = (m&1)? -1.0:1.0;
    q[(l+m)*n + (l+m)] = { s*inv, 0.0 };
    q[(l+m)*n + (l-m)] = { 0.0, s*inv };
  }
  int lm = l&3;
  cplx p = (lm==0)? cplx{1,0} : (lm==1)? cplx{0,-1} : (lm==2)? cplx{-1,0} : cplx{0,1};
  for(int i=0;i<n*n;i++) q[i]=cmul(q[i],p);
}

__global__ void cg_kernel(float* Qtab){
  int t = threadIdx.x;
  int cnt=0;
  for(int a=0;a<3;a++) for(int b=0;b<3;b++){
    int jmin = iabs3(a-b);
    int nj = 2*imin3(a,b)+1;
    int off = qpairOffset(a,b);
    for(int jj=0;jj<nj;jj++){
      int J = jmin+jj;
      int n1=2*a+1, n2=2*J+1, n3=2*b+1;
      if(cnt==t){
        cplx q1[25], q2[81], q3[25];
        r2c_fill(a,q1); r2c_fill(J,q2); r2c_fill(b,q3);
        double C[5][9][5];
        for(int i=0;i<n1;i++)for(int k=0;k<n2;k++)for(int m=0;m<n3;m++)
          C[i][k][m] = su2cg(a,i-a,J,k-J,b,m-b);
        double ccre[5][9][5], ccim[5][9][5];
        double sre=0.0, sim=0.0;
        for(int j=0;j<n1;j++)for(int lI=0;lI<n2;lI++)for(int nI=0;nI<n3;nI++){
          double are=0.0, aim=0.0;
          for(int i=0;i<n1;i++)for(int k=0;k<n2;k++)for(int m=0;m<n3;m++){
            double c = C[i][k][m];
            if(c==0.0) continue;
            cplx term = cmul(cmul(q1[i*n1+j], q2[k*n2+lI]), conjc(q3[m*n3+nI]));
            are += term.re*c; aim += term.im*c;
          }
          ccre[j][lI][nI]=are; ccim[j][lI][nI]=aim;
          sre += fabs(are); sim += fabs(aim);
        }
        bool useRe = (sre>=sim);
        for(int mI=0;mI<n2;mI++)for(int o=0;o<n3;o++)for(int i=0;i<n1;i++)
          Qtab[off + (mI*n3+o)*n1 + i] = (float)(useRe? ccre[i][mI][o] : ccim[i][mI][o]);
      }
      cnt++; off += n1*n2*n3;
    }
  }
}

// ------------------------- edge geometry: Y(l<=4), rad_in -------------------------
__device__ inline float ffact(int n){ float r=1.f; for(int i=2;i<=n;i++) r*=(float)i; return r; }

__global__ void edge_geom_kernel(const float* __restrict__ coords, const float* __restrict__ efeat,
                                 const int* __restrict__ src, const int* __restrict__ dst,
                                 float* __restrict__ Y, float* __restrict__ radin){
  int e = blockIdx.x*blockDim.x + threadIdx.x;
  if(e>=N_EDGES) return;
  int s=src[e], d=dst[e];
  float rx=coords[d*3+0]-coords[s*3+0];
  float ry=coords[d*3+1]-coords[s*3+1];
  float rz=coords[d*3+2]-coords[s*3+2];
  float r = sqrtf(rx*rx+ry*ry+rz*rz);
  float inv = 1.f/(r+1e-8f);
  float ux=rx*inv, uy=ry*inv, uz=rz*inv;
  float ct=uz;
  float st = sqrtf(fminf(fmaxf(1.f-ct*ct,1e-12f),1.f));
  float phi = atan2f(uy,ux);
  float P[5][5];
  P[0][0]=1.f;
  for(int m=1;m<=4;m++) P[m][m] = -(float)(2*m-1)*st*P[m-1][m-1];
  for(int m=0;m<4;m++)  P[m+1][m] = (float)(2*m+1)*ct*P[m][m];
  for(int l=2;l<=4;l++) for(int m=0;m<=l-2;m++)
    P[l][m] = ((float)(2*l-1)*ct*P[l-1][m]-(float)(l+m-1)*P[l-2][m])/(float)(l-m);
  const float PI4 = 12.566370614359172f;
  const float SQ2 = 1.41421356237f;
  float* Ye = Y + (size_t)e*25;
  for(int l=0;l<=4;l++){
    for(int m=-l;m<=l;m++){
      int am = m<0? -m:m;
      float Nc = sqrtf((float)(2*l+1)/PI4 * ffact(l-am)/ffact(l+am));
      float v;
      if(m<0)       v = SQ2*Nc*P[l][am]*sinf((float)am*phi);
      else if(m==0) v = Nc*P[l][0];
      else          v = SQ2*Nc*P[l][am]*cosf((float)m*phi);
      Ye[l*l + (l+m)] = v;
    }
  }
  radin[(size_t)e*5+0]=r;
  for(int t=0;t<4;t++) radin[(size_t)e*5+1+t]=efeat[(size_t)e*4+t];
}

// ------------------------- radial MLP via WMMA -------------------------
__device__ inline int kmapf(int g,int h){ int r=h>>1, o=h&1; return ((r<4)? (2*r) : (16+2*(r-4))) + 8*g + o; }
union AF { v16h v; f16 h[16]; };
union DF { v8f  v; float f[8]; };

#define MLP_BLOCKS 250

__global__ void __launch_bounds__(256) radial_mlp_kernel(
    const float* __restrict__ radin, const float* __restrict__ b1,
    const float* __restrict__ b2, const float* __restrict__ w1,
    const float* __restrict__ w2, const float* __restrict__ w3,
    int OUT, float* __restrict__ Rout)
{
  __shared__ float sw1[5*HID];
  __shared__ float sb1[HID];
  __shared__ float sb2[HID];
  __shared__ f16   sH2[8][16*HID];
  for(int i=threadIdx.x;i<5*HID;i+=blockDim.x) sw1[i]=w1[i];
  for(int i=threadIdx.x;i<HID;i+=blockDim.x){ sb1[i]=b1[i]; sb2[i]=b2[i]; }
  __syncthreads();
  int lane = threadIdx.x & 31;
  int wave = threadIdx.x >> 5;
  int g  = lane >> 4;
  int ml = lane & 15;
  int n  = lane & 15;
  AF bf2a, bf2b, bf3;                        // B fragments (weights), held for whole kernel
  for(int h=0;h<16;h++){
    int k = kmapf(g,h);
    bf2a.h[h] = (f16)w2[k*HID + n];
    bf2b.h[h] = (f16)w2[k*HID + n+16];
    bf3.h[h]  = (n<OUT)? (f16)w3[k*OUT + n] : (f16)0.f;
  }
  const int tiles = (N_EDGES + 15)/16;
  const int wavesTot = MLP_BLOCKS*8;
  const int iters = (tiles + wavesTot - 1)/wavesTot;   // uniform over all blocks
  for(int it=0; it<iters; ++it){
    int tile = it*wavesTot + (int)blockIdx.x*8 + wave;
    bool active = tile < tiles;
    int base = tile*16;
    AF a;
    for(int h=0;h<16;h++) a.h[h]=(f16)0.f;
    if(active){
      int e = base + ml;
      float x[5]={0,0,0,0,0};
      if(e < N_EDGES){ const float* xr = radin + (size_t)e*5; for(int t2=0;t2<5;t2++) x[t2]=xr[t2]; }
      for(int h=0;h<16;h++){                 // GEMM1 (K=5) directly into A fragment
        int k = kmapf(g,h);
        float sacc = sb1[k];
        for(int t2=0;t2<5;t2++) sacc += x[t2]*sw1[t2*HID + k];
        a.h[h] = (f16)fmaxf(sacc, 0.f);
      }
    }
    if(active){
      DF d0, d1;
      for(int i=0;i<8;i++){ d0.f[i]=0.f; d1.f[i]=0.f; }
      d0.v = __builtin_amdgcn_wmma_f32_16x16x32_f16(false, a.v, false, bf2a.v, (short)0, d0.v, false, false);
      d1.v = __builtin_amdgcn_wmma_f32_16x16x32_f16(false, a.v, false, bf2b.v, (short)0, d1.v, false, false);
      for(int r=0;r<8;r++){
        int row = 8*g + r;
        sH2[wave][row*HID + n]      = (f16)fmaxf(d0.f[r] + sb2[n],    0.f);
        sH2[wave][row*HID + n + 16] = (f16)fmaxf(d1.f[r] + sb2[n+16], 0.f);
      }
    }
    __syncthreads();
    AF a3;
    for(int h=0;h<16;h++) a3.h[h] = active ? sH2[wave][ml*HID + kmapf(g,h)] : (f16)0.f;
    __syncthreads();
    if(active){
      DF d3; for(int i=0;i<8;i++) d3.f[i]=0.f;
      d3.v = __builtin_amdgcn_wmma_f32_16x16x32_f16(false, a3.v, false, bf3.v, (short)0, d3.v, false, false);
      if(n < OUT){
        for(int r=0;r<8;r++){
          int e = base + 8*g + r;
          if(e < N_EDGES) Rout[(size_t)e*OUT + n] = d3.f[r];
        }
      }
    }
  }
}

// ------------------------- per-edge CG message kernels -------------------------
// MODE 0: accumulate per-edge value message into vbuf
// MODE 1: accumulate attention score (dot with q[dst]) into score buffer
// MODE 2: atomic segment-sum into per-node out1 (final conv, mo==1)
template<int DI,int DO,int MODE>
__global__ void __launch_bounds__(256) msg_kernel(
  const int* __restrict__ src, const int* __restrict__ dst,
  const float* __restrict__ R, const float* __restrict__ Y,
  const float* __restrict__ hdi, const float* __restrict__ Qtab,
  float* __restrict__ out, const float* __restrict__ qdst, float scale)
{
  constexpr int MI  = MULTD(DI);
  constexpr int MO  = (MODE==2)? 1 : MULTD(DO);
  constexpr int DID = 2*DI+1, DOD = 2*DO+1;
  constexpr int JMIN = (DI>DO)?(DI-DO):(DO-DI);
  constexpr int NJ   = 2*((DI<DO)?DI:DO)+1;
  constexpr int OUT  = MO*MI*NJ;
  constexpr int QLEN = NJ*(2*JMIN+NJ)*DOD*DID;
  __shared__ float Qsh[QLEN];
  const float* Qp = Qtab + qpairOffset(DI,DO);
  for(int i=threadIdx.x;i<QLEN;i+=(int)blockDim.x) Qsh[i]=Qp[i];
  __syncthreads();
  int e = blockIdx.x*blockDim.x + threadIdx.x;
  if(e>=N_EDGES) return;
  int s = src[e];
  const float* hp = hdi + (size_t)s*(MI*DID);
  __builtin_prefetch(hp, 0, 0);
  float hl[MI*DID];
  for(int i=0;i<MI*DID;i++) hl[i]=hp[i];
  float rl[OUT];
  const float* rp = R + (size_t)e*OUT;
  for(int i=0;i<OUT;i++) rl[i]=rp[i];
  const float* Ye = Y + (size_t)e*25;
  float msg[MO*DOD];
  for(int i=0;i<MO*DOD;i++) msg[i]=0.f;
  int qoff=0;
  for(int j=0;j<NJ;j++){
    int J=JMIN+j, tj=2*J+1;
    float bji[DOD*DID];
    for(int i=0;i<DOD*DID;i++) bji[i]=0.f;
    for(int m=0;m<tj;m++){
      float yv = Ye[J*J + m];
      const float* qrow = &Qsh[qoff + m*DOD*DID];
      for(int o=0;o<DOD;o++) for(int i=0;i<DID;i++) bji[o*DID+i] += yv*qrow[o*DID+i];
    }
    qoff += tj*DOD*DID;
    for(int ii=0; ii<MI; ii++){
      for(int o=0;o<DOD;o++){
        float hb=0.f;
        for(int i=0;i<DID;i++) hb += bji[o*DID+i]*hl[ii*DID+i];
        for(int c=0;c<MO;c++) msg[c*DOD+o] += rl[c*MI*NJ + ii*NJ + j]*hb;
      }
    }
  }
  if(MODE==0){
    float* op = out + (size_t)e*(MO*DOD);
    for(int i=0;i<MO*DOD;i++) op[i] += msg[i];
  } else if(MODE==1){
    int d2 = dst[e];
    const float* qd = qdst + (size_t)d2*(MO*DOD);
    float sc=0.f;
    for(int i=0;i<MO*DOD;i++) sc += msg[i]*qd[i];
    out[e] += scale*sc;
  } else {
    int d2 = dst[e];
    for(int o=0;o<DOD;o++) atomicAdd(&out[(size_t)d2*DOD + o], msg[o]);
  }
}

// ------------------------- attention + node kernels -------------------------
__device__ inline unsigned flipf(float x){ unsigned u=__float_as_uint(x); return (u&0x80000000u)? ~u : (u|0x80000000u); }
__device__ inline float unflipf(unsigned u){ unsigned v = (u&0x80000000u)? (u&0x7fffffffu) : ~u; return __uint_as_float(v); }

__global__ void fill0_kernel(float* p, size_t n){
  size_t i = (size_t)blockIdx.x*blockDim.x + threadIdx.x;
  if(i<n) p[i]=0.f;
}

__global__ void qnode_kernel(const float* __restrict__ Wq, const float* __restrict__ h,
                             float* __restrict__ q, int mo, int mi, int dim){
  int n = blockIdx.x*blockDim.x + threadIdx.x;
  if(n>=N_NODES) return;
  for(int c=0;c<mo;c++) for(int m=0;m<dim;m++){
    float s=0.f;
    for(int i=0;i<mi;i++) s += Wq[c*mi+i]*h[(size_t)n*mi*dim + i*dim + m];
    q[(size_t)n*mo*dim + c*dim + m] = s;
  }
}

__global__ void segmax_kernel(const float* __restrict__ score, const int* __restrict__ dst, unsigned* __restrict__ nmax){
  int e = blockIdx.x*blockDim.x + threadIdx.x;
  if(e>=N_EDGES) return;
  atomicMax(&nmax[dst[e]], flipf(score[e]));
}

__global__ void expsum_kernel(const float* __restrict__ score, const int* __restrict__ dst,
                              const unsigned* __restrict__ nmax, float* __restrict__ ex, float* __restrict__ den){
  int e = blockIdx.x*blockDim.x + threadIdx.x;
  if(e>=N_EDGES) return;
  int d = dst[e];
  float v = expf(score[e]-unflipf(nmax[d]));
  ex[e]=v;
  atomicAdd(&den[d], v);
}

__global__ void agg_kernel(const float* __restrict__ ex, const float* __restrict__ den, const int* __restrict__ dst,
                           const float* __restrict__ v0, const float* __restrict__ v1, const float* __restrict__ v2,
                           float* __restrict__ a0, float* __restrict__ a1, float* __restrict__ a2){
  int e = blockIdx.x*blockDim.x + threadIdx.x;
  if(e>=N_EDGES) return;
  int d = dst[e];
  float a = ex[e]/(den[d]+1e-9f);
  for(int i=0;i<4;i++) atomicAdd(&a0[(size_t)d*4+i], a*v0[(size_t)e*4+i]);
  for(int i=0;i<6;i++) atomicAdd(&a1[(size_t)d*6+i], a*v1[(size_t)e*6+i]);
  for(int i=0;i<5;i++) atomicAdd(&a2[(size_t)d*5+i], a*v2[(size_t)e*5+i]);
}

struct DegP {
  const float* agg; const float* hold; const float* proj; const float* skipW;
  const float* normw; const float* normb; float* hnew; int multIn;
};

__global__ void node_update_kernel(DegP d0, DegP d1, DegP d2){
  int n = blockIdx.x*blockDim.x + threadIdx.x;
  if(n>=N_NODES) return;
  DegP dp[3]={d0,d1,d2};
  for(int d=0; d<3; d++){
    int dim=2*d+1, mo=MULTD(d);
    float z[20];
    for(int c=0;c<mo;c++) for(int o=0;o<dim;o++){
      float zz=0.f;
      for(int c2=0;c2<mo;c2++) zz += dp[d].proj[c*mo+c2]*dp[d].agg[(size_t)n*mo*dim + c2*dim + o];
      if(dp[d].skipW){
        int mi=dp[d].multIn;
        for(int c2=0;c2<mi;c2++) zz += dp[d].skipW[c*mi+c2]*dp[d].hold[(size_t)n*mi*dim + c2*dim + o];
      }
      z[c*dim+o]=zz;
    }
    float nm[4], sc[4];
    for(int c=0;c<mo;c++){ float s=0.f; for(int o=0;o<dim;o++) s+=z[c*dim+o]*z[c*dim+o]; nm[c]=sqrtf(s); }
    for(int c2=0;c2<mo;c2++){ float s=dp[d].normb[c2]; for(int c=0;c<mo;c++) s+=nm[c]*dp[d].normw[c*mo+c2]; sc[c2]=fmaxf(s,0.f); }
    for(int c=0;c<mo;c++){ float f=sc[c]/(nm[c]+1e-8f); for(int o=0;o<dim;o++) dp[d].hnew[(size_t)n*mo*dim + c*dim+o] = z[c*dim+o]*f; }
  }
}

__global__ void selfpool_kernel(const float* __restrict__ out1, const float* __restrict__ selfW,
                                const float* __restrict__ h1, float* __restrict__ pooled){
  int n = blockIdx.x*blockDim.x + threadIdx.x;
  if(n>=N_NODES) return;
  for(int o=0;o<3;o++){
    float v = out1[(size_t)n*3+o];
    for(int i=0;i<2;i++) v += selfW[i]*h1[(size_t)n*6 + i*3 + o];
    atomicAdd(&pooled[o], v);
  }
}

__global__ void fc_kernel(const float* pooled, const float* w1, const float* b1,
                          const float* w2, const float* b2, float* outp){
  if(threadIdx.x==0 && blockIdx.x==0){
    float p[3], z[3];
    for(int i=0;i<3;i++) p[i]=pooled[i]/(float)N_NODES;
    for(int o=0;o<3;o++) z[o]=fmaxf(p[0]*w1[0*3+o]+p[1]*w1[1*3+o]+p[2]*w1[2*3+o]+b1[o],0.f);
    outp[0] = z[0]*w2[0]+z[1]*w2[1]+z[2]*w2[2]+b2[0];
  }
}

// ------------------------- host side -------------------------
template<int MODE>
static void msgDispatch(int di,int doo, const int* src,const int* dst,const float* R,const float* Y,
                        const float* hdi,const float* Qtab,float* out,const float* qd,float scale, hipStream_t stream){
  dim3 g((N_EDGES+255)/256), b(256);
  switch(di*3+doo){
    case 0: msg_kernel<0,0,MODE><<<g,b,0,stream>>>(src,dst,R,Y,hdi,Qtab,out,qd,scale); break;
    case 1: msg_kernel<0,1,MODE><<<g,b,0,stream>>>(src,dst,R,Y,hdi,Qtab,out,qd,scale); break;
    case 2: msg_kernel<0,2,MODE><<<g,b,0,stream>>>(src,dst,R,Y,hdi,Qtab,out,qd,scale); break;
    case 3: msg_kernel<1,0,MODE><<<g,b,0,stream>>>(src,dst,R,Y,hdi,Qtab,out,qd,scale); break;
    case 4: msg_kernel<1,1,MODE><<<g,b,0,stream>>>(src,dst,R,Y,hdi,Qtab,out,qd,scale); break;
    case 5: msg_kernel<1,2,MODE><<<g,b,0,stream>>>(src,dst,R,Y,hdi,Qtab,out,qd,scale); break;
    case 6: msg_kernel<2,0,MODE><<<g,b,0,stream>>>(src,dst,R,Y,hdi,Qtab,out,qd,scale); break;
    case 7: msg_kernel<2,1,MODE><<<g,b,0,stream>>>(src,dst,R,Y,hdi,Qtab,out,qd,scale); break;
    case 8: msg_kernel<2,2,MODE><<<g,b,0,stream>>>(src,dst,R,Y,hdi,Qtab,out,qd,scale); break;
  }
}

struct Rad { const float *b1,*b2,*w1,*w2,*w3; int out; };

extern "C" void kernel_launch(void* const* d_in, const int* in_sizes, int n_in,
                              void* d_out, int out_size, void* d_ws, size_t ws_size,
                              hipStream_t stream) {
  (void)in_sizes; (void)out_size; (void)ws_size;
  const float* coords = (const float*)d_in[0];
  const float* feats  = (const float*)d_in[1];
  const float* efeat  = (const float*)d_in[2];
  const int*   esrc   = (const int*)d_in[3];
  const int*   edst   = (const int*)d_in[4];

  // ---- parameter leaves in JAX tree_flatten order (sorted dict keys) ----
  bool packed = (n_in <= 6);
  const float* pbase = (const float*)d_in[5];
  size_t poff=0; int cur=5;
  auto leaf = [&](int cnt)->const float*{
    if(packed){ const float* p=pbase+poff; poff+=(size_t)cnt; return p; }
    return (const float*)d_in[cur++];
  };
  auto leafRad = [&](int out)->Rad{
    Rad r; r.b1=leaf(32); r.b2=leaf(32); r.w1=leaf(5*32); r.w2=leaf(32*32); r.w3=leaf(32*out); r.out=out; return r;
  };
  int MULT[3]={4,2,1};
  auto outFor=[&](int di,int doo)->int{ int nj=2*imin3(di,doo)+1; return MULT[doo]*MULT[di]*nj; };

  // 'fc' < 'final' < 'layers'
  const float* fcb1=leaf(3); const float* fcb2=leaf(1);
  const float* fcw1=leaf(9); const float* fcw2=leaf(3);
  // final.conv '0,1','1,1','2,1' then final.self
  int finOut[3] = {4,6,3};
  Rad finConv[3]; for(int di=0;di<3;di++) finConv[di]=leafRad(finOut[di]);
  const float* finSelf = leaf(2);
  // layers: per layer sorted keys: key, norm, proj, query, skip, value
  struct Layer {
    Rad val[3][3]; bool hasVal[3][3];
    Rad key[3][3]; bool hasKey[3][3];
    const float* query[3]; const float* proj[3]; const float* skipW[3];
    const float* normb[3]; const float* normw[3];
  } L[4];
  for(int li=0; li<4; li++){
    Layer& lp = L[li];
    bool first = (li==0);
    for(int a=0;a<3;a++)for(int b=0;b<3;b++){ lp.hasVal[a][b]=false; lp.hasKey[a][b]=false; }
    for(int d=0;d<3;d++){ lp.query[d]=nullptr; lp.skipW[d]=nullptr; }
    for(int di=0;di<3;di++)for(int doo=0;doo<3;doo++){
      bool present = first ? (di==1 && doo==1) : true;
      if(present){ lp.key[di][doo]=leafRad(outFor(di,doo)); lp.hasKey[di][doo]=true; }
    }
    for(int d=0;d<3;d++){ lp.normb[d]=leaf(MULT[d]); lp.normw[d]=leaf(MULT[d]*MULT[d]); }
    for(int d=0;d<3;d++) lp.proj[d]=leaf(MULT[d]*MULT[d]);
    for(int d=0;d<3;d++){ bool present = first ? (d==1) : true; if(present) lp.query[d]=leaf(MULT[d]*MULT[d]); }
    for(int d=0;d<3;d++){ bool present = first ? (d==1) : true; if(present) lp.skipW[d]=leaf(MULT[d]*MULT[d]); }
    for(int di=0;di<3;di++)for(int doo=0;doo<3;doo++){
      bool present = first ? (di==1) : true;
      if(present){ lp.val[di][doo]=leafRad(outFor(di,doo)); lp.hasVal[di][doo]=true; }
    }
  }

  // ---- workspace layout ----
  float* W = (float*)d_ws;
  size_t off=0;
  auto wsalloc=[&](size_t n)->float*{ float* p=W+off; off += (n+3)&~(size_t)3; return p; };
  const size_t E = N_EDGES, NN = N_NODES;
  float* Ybuf  = wsalloc(E*25);
  float* radin = wsalloc(E*5);
  float* Rbuf  = wsalloc(E*16);
  float* vb[3] = { wsalloc(E*4), wsalloc(E*6), wsalloc(E*5) };
  float* score = wsalloc(E);
  float* exb   = wsalloc(E);
  unsigned* nmax = (unsigned*)wsalloc(NN);
  float* nden  = wsalloc(NN);
  float* qb[3]   = { wsalloc(NN*4), wsalloc(NN*6), wsalloc(NN*5) };
  float* aggb[3] = { wsalloc(NN*4), wsalloc(NN*6), wsalloc(NN*5) };
  float* hA[3]   = { wsalloc(NN*4), wsalloc(NN*6), wsalloc(NN*5) };
  float* hB[3]   = { wsalloc(NN*4), wsalloc(NN*6), wsalloc(NN*5) };
  float* out1    = wsalloc(NN*3);
  float* pooled  = wsalloc(8);
  float* Qtab    = wsalloc(1232);

  dim3 EB((N_EDGES+255)/256), NB((N_NODES+255)/256), TB(256);
  auto fill0=[&](float* p, size_t n){ fill0_kernel<<<dim3((unsigned)((n+255)/256)),TB,0,stream>>>(p,n); };

  cg_kernel<<<1,32,0,stream>>>(Qtab);
  edge_geom_kernel<<<EB,TB,0,stream>>>(coords,efeat,esrc,edst,Ybuf,radin);

  const float* hcur[3] = {nullptr, feats, nullptr};
  float* hnext[3] = {hA[0],hA[1],hA[2]};
  int finMult[3] = {0,2,0};                       // input multiplicities (layer 0: only deg-1, mult 2)

  for(int li=0; li<4; li++){
    Layer& lp = L[li];
    bool first = (li==0);
    fill0(vb[0], E*4); fill0(vb[1], E*6); fill0(vb[2], E*5);
    fill0(score, E);
    fill0((float*)nmax, NN); fill0(nden, NN);
    fill0(aggb[0], NN*4); fill0(aggb[1], NN*6); fill0(aggb[2], NN*5);
    // queries for kdeg degrees
    for(int d=0; d<3; d++){
      if(lp.query[d])
        qnode_kernel<<<NB,TB,0,stream>>>(lp.query[d], hcur[d], qb[d], MULT[d], finMult[d], 2*d+1);
    }
    float invsq = first ? (1.f/sqrtf(6.f)) : (1.f/sqrtf(15.f));
    // value branch
    for(int di=0;di<3;di++) for(int doo=0;doo<3;doo++){
      if(!lp.hasVal[di][doo]) continue;
      Rad& r = lp.val[di][doo];
      radial_mlp_kernel<<<MLP_BLOCKS,TB,0,stream>>>(radin, r.b1,r.b2,r.w1,r.w2,r.w3, r.out, Rbuf);
      msgDispatch<0>(di,doo, esrc,edst,Rbuf,Ybuf,hcur[di],Qtab, vb[doo], nullptr, 0.f, stream);
    }
    // key branch (score accumulation)
    for(int di=0;di<3;di++) for(int doo=0;doo<3;doo++){
      if(!lp.hasKey[di][doo]) continue;
      Rad& r = lp.key[di][doo];
      radial_mlp_kernel<<<MLP_BLOCKS,TB,0,stream>>>(radin, r.b1,r.b2,r.w1,r.w2,r.w3, r.out, Rbuf);
      msgDispatch<1>(di,doo, esrc,edst,Rbuf,Ybuf,hcur[di],Qtab, score, qb[doo], invsq, stream);
    }
    // edge softmax + attention-weighted aggregation
    segmax_kernel<<<EB,TB,0,stream>>>(score,edst,nmax);
    expsum_kernel<<<EB,TB,0,stream>>>(score,edst,nmax,exb,nden);
    agg_kernel<<<EB,TB,0,stream>>>(exb,nden,edst,vb[0],vb[1],vb[2],aggb[0],aggb[1],aggb[2]);
    // node update
    DegP dp[3];
    for(int d=0; d<3; d++){
      dp[d].agg = aggb[d]; dp[d].hold = hcur[d];
      dp[d].proj = lp.proj[d]; dp[d].skipW = lp.skipW[d];
      dp[d].normw = lp.normw[d]; dp[d].normb = lp.normb[d];
      dp[d].hnew = hnext[d]; dp[d].multIn = finMult[d];
    }
    node_update_kernel<<<NB,TB,0,stream>>>(dp[0],dp[1],dp[2]);
    // rotate feature buffers
    for(int d=0; d<3; d++) hcur[d]=hnext[d];
    float** nb = (li&1)? hA : hB;
    for(int d=0; d<3; d++) hnext[d]=nb[d];
    finMult[0]=4; finMult[1]=2; finMult[2]=1;
  }

  // final convolution to degree-1 (no attention) + self-interaction + pooling + FC head
  fill0(out1, NN*3); fill0(pooled, 8);
  for(int di=0; di<3; di++){
    Rad& r = finConv[di];
    radial_mlp_kernel<<<MLP_BLOCKS,TB,0,stream>>>(radin, r.b1,r.b2,r.w1,r.w2,r.w3, r.out, Rbuf);
    msgDispatch<2>(di,1, esrc,edst,Rbuf,Ybuf,hcur[di],Qtab, out1, nullptr, 0.f, stream);
  }
  selfpool_kernel<<<NB,TB,0,stream>>>(out1, finSelf, hcur[1], pooled);
  fc_kernel<<<1,1,0,stream>>>(pooled, fcw1, fcb1, fcw2, fcb2, (float*)d_out);
}
